// MCDGnn_4896262717832
// MI455X (gfx1250) — matrix operand order repaired and checked by reading, not compiled, not run on hardware
//
#include <hip/hip_runtime.h>
#include <hip/hip_bf16.h>
#include <math.h>

// ---------- WMMA fragment types (CDNA5 wave32) ----------
typedef __attribute__((ext_vector_type(16))) __bf16 v16bf;
typedef __attribute__((ext_vector_type(8)))  float  v8f;
typedef __attribute__((ext_vector_type(4)))  float  v4f_;
typedef __attribute__((ext_vector_type(4)))  unsigned int v4u_;
typedef __attribute__((ext_vector_type(8)))  int    v8i_;
typedef __attribute__((ext_vector_type(4)))  int    v4i_;

union Frag16 { unsigned int u[8]; v16bf v; };

__device__ __forceinline__ unsigned short f2bf(float f) {
  unsigned int u = __float_as_uint(f);
  u += 0x7FFFu + ((u >> 16) & 1u);           // round-to-nearest-even
  return (unsigned short)(u >> 16);
}
__device__ __forceinline__ unsigned int pk2bf(float a, float b) {
  return (unsigned int)f2bf(a) | ((unsigned int)f2bf(b) << 16);
}

#define KDIM 1024
#define HID  128

#if __has_builtin(__builtin_amdgcn_tensor_load_to_lds) && __has_builtin(__builtin_amdgcn_s_wait_tensorcnt)
#define USE_TDM 1
#else
#define USE_TDM 0
#endif

#if USE_TDM
// TDM: DMA one 32(k) x 128(col) bf16 tile of Wt into LDS at lds_addr.
// D# per CDNA5 ISA: 2D tensor, data_size=2B,
// tensor_dim0=1024 (k extent), dim0_stride=1024, tile 32x128.
__device__ __forceinline__ void tdm_load_B(unsigned lds_addr, const unsigned short* gsrc) {
  unsigned long long ga = (unsigned long long)(uintptr_t)gsrc;
  v4u_ g0;
  g0[0] = 1u;                                            // count=1 (valid user D#)
  g0[1] = lds_addr;                                      // lds_addr [63:32]
  g0[2] = (unsigned)(ga & 0xFFFFFFFFu);                  // global_addr lo
  g0[3] = (unsigned)((ga >> 32) & 0x01FFFFFFu) | 0x80000000u; // global_addr hi | type=2
  v8i_ g1;
  g1[0] = 0x00010000;                                    // wg_mask=0, data_size=1(2B)
  g1[1] = (int)((KDIM & 0xFFFFu) << 16);                 // tensor_dim0 lo16 << 16
  g1[2] = (int)((KDIM >> 16) | ((HID & 0xFFFFu) << 16)); // dim0 hi16 | tensor_dim1 lo16
  g1[3] = (int)(32u << 16);                              // dim1 hi16=0 | tile_dim0=32
  g1[4] = HID;                                           // tile_dim1=128, tile_dim2=0
  g1[5] = KDIM;                                          // tensor_dim0_stride lo32
  g1[6] = 0;                                             // stride hi16 | dim1_stride lo
  g1[7] = 0;
  v4i_ z4 = {0, 0, 0, 0};
#if defined(__clang_major__) && (__clang_major__ >= 23)
  v8i_ z8 = {0, 0, 0, 0, 0, 0, 0, 0};
  __builtin_amdgcn_tensor_load_to_lds(g0, g1, z4, z4, z8, 0);
#else
  __builtin_amdgcn_tensor_load_to_lds(g0, g1, z4, z4, 0);
#endif
}
#endif

// ---------- K0: init scratch + output ----------
__global__ void k_init(float* __restrict__ outacc, float* __restrict__ m,
                       float* __restrict__ denom, float* __restrict__ dout,
                       long accN, int N, int G) {
  long i = (long)blockIdx.x * blockDim.x + threadIdx.x;
  if (i < accN) outacc[i] = 0.0f;
  if (i < N) { m[i] = __int_as_float(0xFF800000); denom[i] = 0.0f; }
  if (i < (long)G * HID) dout[i] = 0.0f;
}

// ---------- K0b: W [1024x128] f32 -> Wt [128][1024] bf16 (transposed) ----------
__global__ void k_wt(const float* __restrict__ W, unsigned short* __restrict__ Wt) {
  int i = blockIdx.x * blockDim.x + threadIdx.x;   // i < 1024*128
  int k = i >> 7, n = i & 127;
  Wt[n * KDIM + k] = f2bf(W[i]);
}

// ---------- K1: h = x @ W via bf16 WMMA, 128x128 block tile ----------
__global__ __launch_bounds__(256) void k_gemm(const float* __restrict__ x,
                                              const unsigned short* __restrict__ Wt,
                                              float* __restrict__ h, int M) {
  __shared__ unsigned short Ab[2][128 * 32];   // [row][k]   8KB each
  __shared__ unsigned short Bb[2][128 * 32];   // [col][k]   8KB each

  const int t = threadIdx.x;
  const int w = t >> 5, lane = t & 31;
  const int rbase = blockIdx.x * 128;

  v8f acc[8];
#pragma unroll
  for (int i = 0; i < 8; ++i)
#pragma unroll
    for (int j = 0; j < 8; ++j) acc[i][j] = 0.0f;

  // A tile staging: rows rbase..+127, k kc..kc+31, f32 -> bf16 inline.
  // x is streamed exactly once -> non-temporal loads keep L2 for h/Wt.
  auto stageA = [&](int buf, int kc) {
    int row = t >> 1, half = t & 1;
    long grow = (long)rbase + row;
    const v4f_* src = (const v4f_*)(x + grow * KDIM + kc + half * 16);
    unsigned int* dst = (unsigned int*)&Ab[buf][row * 32 + half * 16];
#pragma unroll
    for (int j = 0; j < 4; ++j) {
      v4f_ f = {0.f, 0.f, 0.f, 0.f};
      if (grow < M) f = __builtin_nontemporal_load(src + j);
      dst[j * 2 + 0] = pk2bf(f.x, f.y);
      dst[j * 2 + 1] = pk2bf(f.z, f.w);
    }
  };
#if !USE_TDM
  auto stageB = [&](int buf, int kc) {
    int col = t >> 1, half = t & 1;
    const uint4* src = (const uint4*)(Wt + (long)col * KDIM + kc + half * 16);
    uint4* dst = (uint4*)&Bb[buf][col * 32 + half * 16];
    dst[0] = src[0];
    dst[1] = src[1];
  };
#endif

  stageA(0, 0);
#if USE_TDM
  if (w == 0)
    tdm_load_B((unsigned)(uintptr_t)&Bb[0][0], Wt);
  __builtin_amdgcn_s_wait_tensorcnt(0);
#else
  stageB(0, 0);
#endif
  __syncthreads();

  const int NSTEP = KDIM / 32;   // 32
#pragma unroll 1
  for (int s = 0; s < NSTEP; ++s) {
    const int cur = s & 1;
    if (s + 1 < NSTEP) {
      stageA(cur ^ 1, (s + 1) * 32);
#if USE_TDM
      if (w == 0)
        tdm_load_B((unsigned)(uintptr_t)&Bb[cur ^ 1][0], Wt + (s + 1) * 32);
#else
      stageB(cur ^ 1, (s + 1) * 32);
#endif
      if (s + 2 < NSTEP) {   // global_prefetch_b8 for the tile after next
        const char* pf = (const char*)(x + ((long)rbase + (t >> 1)) * KDIM + (s + 2) * 32);
        __builtin_prefetch(pf, 0, 0);
      }
    }

    // A fragment: 16x32 bf16, rows w*16..+15
    Frag16 fa;
    {
      const unsigned int* Arow =
          (const unsigned int*)&Ab[cur][(w * 16 + (lane & 15)) * 32];
      const int kb2 = (lane < 16) ? 0 : 4;   // lane<16: K0..7/16..23; lane>=16: K8..15/24..31
      uint4 lo = *(const uint4*)(Arow + kb2);
      uint4 hi = *(const uint4*)(Arow + kb2 + 8);
      fa.u[0] = lo.x; fa.u[1] = lo.y; fa.u[2] = lo.z; fa.u[3] = lo.w;
      fa.u[4] = hi.x; fa.u[5] = hi.y; fa.u[6] = hi.z; fa.u[7] = hi.w;
    }

    // Load ALL 8 B fragments first so the 8 WMMAs issue back-to-back.
    Frag16 fb[8];
#pragma unroll
    for (int n0 = 0; n0 < 8; ++n0) {
      const unsigned int* Bcol =
          (const unsigned int*)&Bb[cur][(n0 * 16 + (lane & 15)) * 32];
      const int o = (lane < 16) ? 0 : 4;     // lane<16: K0..7/16..23; lane>=16: K8..15/24..31
      uint4 blo = *(const uint4*)(Bcol + o);
      uint4 bhi = *(const uint4*)(Bcol + o + 8);
      fb[n0].u[0] = blo.x; fb[n0].u[1] = blo.y; fb[n0].u[2] = blo.z; fb[n0].u[3] = blo.w;
      fb[n0].u[4] = bhi.x; fb[n0].u[5] = bhi.y; fb[n0].u[6] = bhi.z; fb[n0].u[7] = bhi.w;
    }
#pragma unroll
    for (int n0 = 0; n0 < 8; ++n0)
      acc[n0] = __builtin_amdgcn_wmma_f32_16x16x32_bf16(
          false, fa.v, false, fb[n0].v, (short)0, acc[n0], false, false);

#if USE_TDM
    __builtin_amdgcn_s_wait_tensorcnt(0);   // wave0's pending TDM (others: nop)
#endif
    __syncthreads();
  }

  // Epilogue: C layout — lane<16: VGPR j = row j, N=lane; lane>=16: row j+8, N=lane-16
  const int rb = rbase + w * 16;
  const int col0 = lane & 15;
  const int mo = (lane < 16) ? 0 : 8;
#pragma unroll
  for (int n0 = 0; n0 < 8; ++n0)
#pragma unroll
    for (int j = 0; j < 8; ++j) {
      int row = rb + mo + j;
      if (row < M) h[(long)row * HID + n0 * 16 + col0] = acc[n0][j];
    }
}

// ---------- K2: a_src = h@att_src, a_dst = h@att_dst (wave per row) ----------
__global__ __launch_bounds__(256) void k_scores(const float* __restrict__ h,
                                                const float* __restrict__ att_s,
                                                const float* __restrict__ att_d,
                                                float* __restrict__ a_src,
                                                float* __restrict__ a_dst, int N) {
  int w = threadIdx.x >> 5, lane = threadIdx.x & 31;
  int row = blockIdx.x * 8 + w;
  if (row >= N) return;
  float4 hv = ((const float4*)(h + (long)row * HID))[lane];
  float4 as = ((const float4*)att_s)[lane];
  float4 ad = ((const float4*)att_d)[lane];
  float ps = hv.x * as.x + hv.y * as.y + hv.z * as.z + hv.w * as.w;
  float pd = hv.x * ad.x + hv.y * ad.y + hv.z * ad.z + hv.w * ad.w;
#pragma unroll
  for (int off = 16; off > 0; off >>= 1) {
    ps += __shfl_down(ps, off, 32);
    pd += __shfl_down(pd, off, 32);
  }
  if (lane == 0) { a_src[row] = ps; a_dst[row] = pd; }
}

// ---------- K3: edge scores + segment max over dst ----------
__global__ void k_edge_max(const int* __restrict__ ei, const float* __restrict__ a_src,
                           const float* __restrict__ a_dst, float* __restrict__ score,
                           float* __restrict__ m, int E, int N) {
  long i = (long)blockIdx.x * blockDim.x + threadIdx.x;
  long T = (long)E + N;
  if (i >= T) return;
  int s, d;
  if (i < E) { s = ei[i]; d = ei[E + i]; } else { s = d = (int)(i - E); }
  float sc = a_src[s] + a_dst[d];
  sc = sc > 0.0f ? sc : 0.2f * sc;          // leaky_relu(0.2)
  score[i] = sc;
  // float atomic-max via ordered int/uint trick (m init = -inf)
  if (sc >= 0.0f) atomicMax((int*)(m + d), __float_as_int(sc));
  else            atomicMin((unsigned int*)(m + d), __float_as_uint(sc));
}

// ---------- K4: p = exp(score - m[dst]); denom[dst] += p ----------
__global__ void k_edge_sum(const int* __restrict__ ei, const float* __restrict__ score,
                           const float* __restrict__ m, float* __restrict__ p,
                           float* __restrict__ denom, int E, int N) {
  long i = (long)blockIdx.x * blockDim.x + threadIdx.x;
  long T = (long)E + N;
  if (i >= T) return;
  int d = (i < E) ? ei[E + i] : (int)(i - E);
  float pe = __expf(score[i] - m[d]);
  p[i] = pe;
  unsafeAtomicAdd(denom + d, pe);           // native global_atomic_add_f32
}

// ---------- K5: out[dst] += alpha * h[src]  (wave per edge) ----------
__global__ __launch_bounds__(256) void k_aggregate(const int* __restrict__ ei,
                                                   const float* __restrict__ p,
                                                   const float* __restrict__ denom,
                                                   const float* __restrict__ h,
                                                   float* __restrict__ outacc,
                                                   int E, int N) {
  int w = threadIdx.x >> 5, lane = threadIdx.x & 31;
  long e = (long)blockIdx.x * 8 + w;
  long T = (long)E + N;
  if (e >= T) return;
  int s, d;
  if (e < E) { s = ei[e]; d = ei[E + e]; } else { s = d = (int)(e - E); }
  float alpha = p[e] / denom[d];
  float4 hv = ((const float4*)(h + (long)s * HID))[lane];  // L2-resident gather
  float* dst = outacc + (long)d * HID + lane * 4;
  unsafeAtomicAdd(dst + 0, alpha * hv.x);
  unsafeAtomicAdd(dst + 1, alpha * hv.y);
  unsafeAtomicAdd(dst + 2, alpha * hv.z);
  unsafeAtomicAdd(dst + 3, alpha * hv.w);
}

// ---------- K6: relu(out+bias), then global max-pool per graph ----------
__global__ __launch_bounds__(256) void k_pool(const float* __restrict__ outacc,
                                              const float* __restrict__ bias,
                                              const int* __restrict__ batch,
                                              float* __restrict__ out, int N) {
  int w = threadIdx.x >> 5, lane = threadIdx.x & 31;
  int row = blockIdx.x * 8 + w;
  if (row >= N) return;
  float4 v = ((const float4*)(outacc + (long)row * HID))[lane];
  float4 b = ((const float4*)bias)[lane];
  float r0 = fmaxf(v.x + b.x, 0.0f);
  float r1 = fmaxf(v.y + b.y, 0.0f);
  float r2 = fmaxf(v.z + b.z, 0.0f);
  float r3 = fmaxf(v.w + b.w, 0.0f);
  int g = batch[row];
  int* o = (int*)out + (long)g * HID + lane * 4;   // relu>=0 -> int-ordered max valid
  atomicMax(o + 0, __float_as_int(r0));
  atomicMax(o + 1, __float_as_int(r1));
  atomicMax(o + 2, __float_as_int(r2));
  atomicMax(o + 3, __float_as_int(r3));
}

// ---------- host launcher ----------
extern "C" void kernel_launch(void* const* d_in, const int* in_sizes, int n_in,
                              void* d_out, int out_size, void* d_ws, size_t ws_size,
                              hipStream_t stream) {
  const float* x       = (const float*)d_in[0];
  const float* W       = (const float*)d_in[1];
  const float* att_src = (const float*)d_in[2];
  const float* att_dst = (const float*)d_in[3];
  const float* bias    = (const float*)d_in[4];
  const int*   ei      = (const int*)d_in[5];
  const int*   batch   = (const int*)d_in[6];

  const int N = in_sizes[6];         // 50000 nodes
  const int E = in_sizes[5] / 2;     // 1.6M edges
  const int G = out_size / HID;      // 64 graphs
  float* out = (float*)d_out;

  // carve workspace (256B aligned)
  char* ws = (char*)d_ws;
  auto carve = [&](size_t bytes) { char* q = ws; ws += (bytes + 255) & ~(size_t)255; return q; };
  float* h      = (float*)carve((size_t)N * HID * 4);
  float* outacc = (float*)carve((size_t)N * HID * 4);
  float* a_src  = (float*)carve((size_t)N * 4);
  float* a_dst  = (float*)carve((size_t)N * 4);
  float* m      = (float*)carve((size_t)N * 4);
  float* denom  = (float*)carve((size_t)N * 4);
  long   T      = (long)E + N;
  float* score  = (float*)carve((size_t)T * 4);
  float* p      = (float*)carve((size_t)T * 4);
  unsigned short* Wt = (unsigned short*)carve((size_t)KDIM * HID * 2);

  long accN = (long)N * HID;
  k_init<<<(unsigned)((accN + 255) / 256), 256, 0, stream>>>(outacc, m, denom, out, accN, N, G);
  k_wt<<<(KDIM * HID) / 256, 256, 0, stream>>>(W, Wt);
  k_gemm<<<(N + 127) / 128, 256, 0, stream>>>(x, Wt, h, N);
  k_scores<<<(N + 7) / 8, 256, 0, stream>>>(h, att_src, att_dst, a_src, a_dst, N);
  k_edge_max<<<(unsigned)((T + 255) / 256), 256, 0, stream>>>(ei, a_src, a_dst, score, m, E, N);
  k_edge_sum<<<(unsigned)((T + 255) / 256), 256, 0, stream>>>(ei, score, m, p, denom, E, N);
  k_aggregate<<<(unsigned)((T + 7) / 8), 256, 0, stream>>>(ei, p, denom, h, outacc, E, N);
  k_pool<<<(N + 7) / 8, 256, 0, stream>>>(outacc, bias, batch, out, N);
}
